// VQEmbedding_39797166964991
// MI455X (gfx1250) — matrix operand order repaired and checked by reading, not compiled, not run on hardware
//
#include <hip/hip_runtime.h>
#include <hip/hip_bf16.h>

// ---------------------------------------------------------------------------
// VQ-VAE codebook lookup on MI455X (gfx1250, wave32, WMMA + TDM).
//
//  * argmin fused into WMMA GEMM epilogue (dist matrix never materialized;
//    that would be 256 MB of HBM traffic vs the ~193 MB compulsory floor).
//  * fp32 GEMM via bf16 hi/lo split: 3x v_wmma_f32_16x16x32_bf16 per chunk.
//  * A panel (16x256 per wave) pinned in VGPRs, reused for all 64 code tiles.
//  * W tiles staged into LDS by the Tensor Data Mover (double-buffered,
//    TENSORcnt + barrier), so the 8 waves/WG share one copy instead of each
//    streaming 1 MB from L2 (4 GB -> 512 MB of L2 traffic).
//  * TDM pad_enable inserts 16B per 512B row -> 528B LDS row stride ->
//    bank-conflict-free ds_load of B fragments.
// ---------------------------------------------------------------------------

typedef __attribute__((ext_vector_type(16))) __bf16 v16bf;
typedef __attribute__((ext_vector_type(8)))  __bf16 v8bf;
typedef __attribute__((ext_vector_type(8)))  float  v8f;
typedef __attribute__((ext_vector_type(4)))  unsigned int u32x4;
typedef __attribute__((ext_vector_type(8)))  int i32x8;
typedef __attribute__((ext_vector_type(4)))  int i32x4;

#define DD   256        // D
#define NN   256        // B == T
#define KK   1024       // codebook entries
#define NROW (NN*NN)    // 65536 flattened (b,t) rows
#define LROW 264        // LDS row stride in bf16 elements (256 + 8 pad)

// ---------------------------------------------------------------------------
// Kernel 1: split W (fp32) into bf16 hi/lo and compute ||W_k||^2.
// ---------------------------------------------------------------------------
__global__ __launch_bounds__(256) void vq_prep_kernel(
    const float* __restrict__ W, __bf16* __restrict__ Whi,
    __bf16* __restrict__ Wlo, float* __restrict__ wsq)
{
  const int k = blockIdx.x;
  const int d = threadIdx.x;
  const float f = W[k * DD + d];
  const __bf16 h = (__bf16)f;
  Whi[k * DD + d] = h;
  Wlo[k * DD + d] = (__bf16)(f - (float)h);

  float s = f * f;
  #pragma unroll
  for (int m = 16; m >= 1; m >>= 1) s += __shfl_xor(s, m, 32);
  __shared__ float part[8];
  if ((threadIdx.x & 31) == 0) part[threadIdx.x >> 5] = s;
  __syncthreads();
  if (threadIdx.x == 0) {
    float t = 0.f;
    #pragma unroll
    for (int i = 0; i < 8; ++i) t += part[i];
    wsq[k] = t;
  }
}

// ---------------------------------------------------------------------------
// TDM: DMA one 8 KB W tile (16 codes x 256 bf16, contiguous in global) into
// LDS at byte offset lds_off, inserting 16B of padding after every 512B row
// (pad_interval=6 -> 128 dwords, pad_amount=3 -> 4 dwords) => 528B row stride.
// D# built per CDNA5 ISA (group0/group1 bitfields);
// this toolchain exposes the 6-arg builtin form.
// ---------------------------------------------------------------------------
__device__ __forceinline__ void tdm_load_tile(const __bf16* gsrc, unsigned int lds_off)
{
  const unsigned long long ga = (unsigned long long)(uintptr_t)gsrc;
  union { unsigned int w[4]; u32x4 v; } g0;
  g0.w[0] = 1u;                                   // count=1, user descriptor
  g0.w[1] = lds_off;                              // lds_addr (bytes)
  g0.w[2] = (unsigned int)ga;                     // global_addr[31:0]
  g0.w[3] = (unsigned int)((ga >> 32) & 0x01FFFFFFu) | 0x80000000u; // [56:32] | type=2

  union { int w[8]; i32x8 v; } g1;
  // data_size=3 (8B), pad_enable, pad_interval=6 (128 dw), pad_amount=3 (4 dw)
  g1.w[0] = (3 << 16) | (1 << 20) | (6 << 22) | (3 << 25);
  g1.w[1] = (int)(1024u << 16);                   // tensor_dim0 = 1024 (8B units)
  g1.w[2] = (int)(1u << 16);                      // tensor_dim1 = 1
  g1.w[3] = (int)(1024u << 16);                   // tile_dim0 = 1024
  g1.w[4] = 1;                                    // tile_dim1 = 1, tile_dim2 = 0
  g1.w[5] = 1024;                                 // tensor_dim0_stride[31:0]
  g1.w[6] = 0;
  g1.w[7] = 0;

  const i32x4 gz4 = {0, 0, 0, 0};
  const i32x8 gz8 = {0, 0, 0, 0, 0, 0, 0, 0};
  __builtin_amdgcn_tensor_load_to_lds(g0.v, g1.v, gz4, gz4, gz8, 0);
}

// ---------------------------------------------------------------------------
// Kernel 2: fused WMMA GEMM (-2 x.W^T + ||W||^2) + argmin over K.
// grid = NROW/128, block = 256 (8 waves x 16 rows). W tiles via TDM->LDS.
// ---------------------------------------------------------------------------
__global__ __launch_bounds__(256) void vq_argmin_kernel(
    const float* __restrict__ x, const __bf16* __restrict__ Whi,
    const __bf16* __restrict__ Wlo, const float* __restrict__ wsq,
    int* __restrict__ ids)
{
  __shared__ __align__(16) __bf16 sW[2][2][16 * LROW];  // [buf][hi/lo][tile]

  const int lane = threadIdx.x & 31;
  const int wave = threadIdx.x >> 5;
  const int r16  = lane & 15;   // A: row in tile; B/C: column in tile
  const int kh   = lane >> 4;   // half-wave select
  const int kbA  = kh * 8;      // unused: A loads with kbB (contiguous order)
  const int kbB  = kh * 16;     // B-matrix K base within 32-chunk

  const int rowBase = blockIdx.x * 128 + wave * 16;
  const float* xrow = x + (size_t)(rowBase + r16) * DD;

  // Load + split this wave's 16x256 A panel into bf16 hi/lo fragments.
  // A 16x32 bf16 layout: element e -> K = kbB + e (contiguous 16).
  v16bf axh[8], axl[8];
  #pragma unroll
  for (int c = 0; c < 8; ++c) {
    const v8f r0 = *(const v8f*)(xrow + c * 32 + kbB);        // K = kbB..kbB+7
    const v8f r1 = *(const v8f*)(xrow + c * 32 + kbB + 8);    // K = kbB+8..+15
    #pragma unroll
    for (int j = 0; j < 8; ++j) {
      const float f0 = r0[j], f1 = r1[j];
      const __bf16 h0 = (__bf16)f0, h1 = (__bf16)f1;
      axh[c][j]     = h0;  axl[c][j]     = (__bf16)(f0 - (float)h0);
      axh[c][j + 8] = h1;  axl[c][j + 8] = (__bf16)(f1 - (float)h1);
    }
  }

  float minv[8];
  int   mini[8];
  #pragma unroll
  for (int j = 0; j < 8; ++j) { minv[j] = 3.4e38f; mini[j] = 0; }

  // Prologue: stage tile 0 into buffer 0.
  if (wave == 0) {
    tdm_load_tile(Whi, (unsigned int)(uintptr_t)&sW[0][0][0]);
    tdm_load_tile(Wlo, (unsigned int)(uintptr_t)&sW[0][1][0]);
    __builtin_amdgcn_s_wait_tensorcnt(0);
  }
  __syncthreads();

  for (int nt = 0; nt < 64; ++nt) {
    const int cur = nt & 1;
    // Kick DMA for the next tile while this one is consumed.
    if (wave == 0 && nt + 1 < 64) {
      const size_t goff = (size_t)(nt + 1) * 16 * DD;
      tdm_load_tile(Whi + goff, (unsigned int)(uintptr_t)&sW[cur ^ 1][0][0]);
      tdm_load_tile(Wlo + goff, (unsigned int)(uintptr_t)&sW[cur ^ 1][1][0]);
    }

    const int cbase = nt * 16;
    const __bf16* lwh = &sW[cur][0][r16 * LROW];
    const __bf16* lwl = &sW[cur][1][r16 * LROW];

    v8f acc = {};
    #pragma unroll
    for (int c = 0; c < 8; ++c) {
      // B 32x16 bf16 layout: lane's column = r16, K = kbB + e (contiguous 16).
      union { v16bf v; v8bf h[2]; } bh, bl;
      bh.h[0] = *(const v8bf*)(lwh + c * 32 + kbB);
      bh.h[1] = *(const v8bf*)(lwh + c * 32 + kbB + 8);
      bl.h[0] = *(const v8bf*)(lwl + c * 32 + kbB);
      bl.h[1] = *(const v8bf*)(lwl + c * 32 + kbB + 8);
      acc = __builtin_amdgcn_wmma_f32_16x16x32_bf16(false, axh[c], false, bh.v,
                                                    (short)0, acc, false, false);
      acc = __builtin_amdgcn_wmma_f32_16x16x32_bf16(false, axh[c], false, bl.v,
                                                    (short)0, acc, false, false);
      acc = __builtin_amdgcn_wmma_f32_16x16x32_bf16(false, axl[c], false, bh.v,
                                                    (short)0, acc, false, false);
    }

    // Epilogue: score = ||W_k||^2 - 2 x.W_k ; fold into running per-row min.
    const float wsqc = wsq[cbase + r16];
    #pragma unroll
    for (int j = 0; j < 8; ++j) {            // C/D: vgpr j -> row j + 8*kh
      const float score = wsqc - 2.0f * acc[j];
      if (score < minv[j]) { minv[j] = score; mini[j] = cbase + r16; }
    }

    // Publish next buffer: DMA complete (issuing wave) + workgroup barrier.
    if (wave == 0 && nt + 1 < 64) __builtin_amdgcn_s_wait_tensorcnt(0);
    __syncthreads();
  }

  // Argmin reduction across the 16 columns (lanes within each half-wave).
  #pragma unroll
  for (int j = 0; j < 8; ++j) {
    float v = minv[j]; int i = mini[j];
    #pragma unroll
    for (int m = 8; m >= 1; m >>= 1) {
      const float ov = __shfl_xor(v, m, 32);
      const int   oi = __shfl_xor(i, m, 32);
      if (ov < v || (ov == v && oi < i)) { v = ov; i = oi; }
    }
    mini[j] = i;
  }
  if (r16 == 0) {
    #pragma unroll
    for (int j = 0; j < 8; ++j) ids[rowBase + kh * 8 + j] = mini[j];
  }
}

// ---------------------------------------------------------------------------
// Kernel 3: losses (commitment == codebook numerically) + qs[b,t].
// ---------------------------------------------------------------------------
__global__ __launch_bounds__(256) void vq_loss_kernel(
    const float* __restrict__ x, const float* __restrict__ W,
    const float* __restrict__ z, const int* __restrict__ ids,
    float* __restrict__ loss_c, float* __restrict__ loss_b,
    float* __restrict__ qs)
{
  const int bt = blockIdx.x;
  const int d  = threadIdx.x;
  const int id = ids[bt];
  const float df = W[id * DD + d] - x[(size_t)bt * DD + d];
  float s = df * df;
  #pragma unroll
  for (int m = 16; m >= 1; m >>= 1) s += __shfl_xor(s, m, 32);
  __shared__ float part[8];
  if ((threadIdx.x & 31) == 0) part[threadIdx.x >> 5] = s;
  __syncthreads();
  if (threadIdx.x == 0) {
    float t = 0.f;
    #pragma unroll
    for (int i = 0; i < 8; ++i) t += part[i];
    t *= (1.0f / (float)DD);
    loss_c[bt] = t;
    loss_b[bt] = t;
    // q[b,t,128] = W[id,128] * sample[...,128] ; sample[...,128] = z[b,t,0]
    qs[bt] = W[id * DD + 128] * z[(size_t)bt * (DD / 2)];
  }
}

// ---------------------------------------------------------------------------
// Kernel 4: qr[t,d] = W[ids[128,t],d] * (d<128 ? 1 : z[128,t,d-128]).
// ---------------------------------------------------------------------------
__global__ __launch_bounds__(256) void vq_qr_kernel(
    const float* __restrict__ W, const float* __restrict__ z,
    const int* __restrict__ ids, float* __restrict__ qr)
{
  const int t = blockIdx.x;
  const int d = threadIdx.x;
  const int id = ids[128 * NN + t];
  const float wv = W[id * DD + d];
  const float sm = (d < 128) ? 1.0f
                             : z[(size_t)(128 * NN + t) * (DD / 2) + (d - 128)];
  qr[t * DD + d] = wv * sm;
}

// ---------------------------------------------------------------------------
// Kernel 5: out[b,t,d] = qs[b,t] + qr[t,d]  (the only 64 MB stream).
// ---------------------------------------------------------------------------
__global__ __launch_bounds__(64) void vq_out_kernel(
    const float* __restrict__ qs, const float* __restrict__ qr,
    float* __restrict__ out)
{
  const int bt = blockIdx.x;
  const int t  = bt & (NN - 1);
  const float s = qs[bt];
  const float4 v = ((const float4*)(qr + t * DD))[threadIdx.x];
  ((float4*)(out + (size_t)bt * DD))[threadIdx.x] =
      make_float4(v.x + s, v.y + s, v.z + s, v.w + s);
}

// ---------------------------------------------------------------------------
extern "C" void kernel_launch(void* const* d_in, const int* in_sizes, int n_in,
                              void* d_out, int out_size, void* d_ws, size_t ws_size,
                              hipStream_t stream) {
  const float* x = (const float*)d_in[0];   // [256,256,256]
  const float* W = (const float*)d_in[1];   // [1024,256]
  const float* z = (const float*)d_in[2];   // [256,256,128]

  // Outputs concatenated flat: out[16777216], ids[65536], loss_c[65536], loss_b[65536]
  float* out    = (float*)d_out;
  int*   ids    = (int*)(out + (size_t)NROW * DD);
  float* loss_c = (float*)(ids + NROW);
  float* loss_b = loss_c + NROW;

  // Workspace layout (~1.58 MB)
  char* ws = (char*)d_ws;
  __bf16* Whi = (__bf16*)ws;                       ws += (size_t)KK * DD * 2;
  __bf16* Wlo = (__bf16*)ws;                       ws += (size_t)KK * DD * 2;
  float*  wsq = (float*)ws;                        ws += KK * 4;
  float*  qs  = (float*)ws;                        ws += NROW * 4;
  float*  qr  = (float*)ws;

  vq_prep_kernel<<<KK, 256, 0, stream>>>(W, Whi, Wlo, wsq);
  vq_argmin_kernel<<<NROW / 128, 256, 0, stream>>>(x, Whi, Wlo, wsq, ids);
  vq_loss_kernel<<<NROW, 256, 0, stream>>>(x, W, z, ids, loss_c, loss_b, qs);
  vq_qr_kernel<<<NN, 256, 0, stream>>>(W, z, ids, qr);
  vq_out_kernel<<<NROW, 64, 0, stream>>>(qs, qr, out);
}
